// RNNScratch_10548439679035
// MI455X (gfx1250) — matrix operand order repaired and checked by reading, compiled, not run on hardware
//
#include <hip/hip_runtime.h>
#include <hip/hip_bf16.h>
#include <math.h>

// Tanh RNN for MI455X (gfx1250, wave32, WMMA).
// T=512, B=64, I=H=1024.
//   Phase 0: convert inputs f32->bf16 (streaming, bandwidth-bound ~9us)
//            + transpose weights to bf16 [N,K]
//   Phase 1: xW[t,b,:] = x[t,b,:] @ W_xh + b_h   (bf16 WMMA GEMM, 4 N-tiles/wave)
//   Phase 2: h_t = tanh(xW_t + h_{t-1} @ W_hh)   (512 sequential step kernels)

#define T_STEPS 512
#define BATCH   64
#define IDIM    1024
#define HDIM    1024

typedef __attribute__((ext_vector_type(16))) __bf16 v16bf;
typedef __attribute__((ext_vector_type(8)))  float  v8f;

__device__ __forceinline__ unsigned short f32_to_bf16_rne(float f) {
  unsigned int u = __float_as_uint(f);
  unsigned int r = u + 0x7FFFu + ((u >> 16) & 1u);   // round-to-nearest-even
  return (unsigned short)(r >> 16);
}

__device__ __forceinline__ unsigned pack_bf16x2(float lo, float hi) {
  return (unsigned)f32_to_bf16_rne(lo) | ((unsigned)f32_to_bf16_rne(hi) << 16);
}

union Frag16 {
  v16bf v;
  uint4 q[2];
  unsigned u[8];
};

// ---------------------------------------------------------------------------
// Streaming f32 -> bf16 conversion, 8 elements per thread (b128 in, b128 out).
// ---------------------------------------------------------------------------
__global__ void k_f32_to_bf16(const float* __restrict__ src,
                              unsigned short* __restrict__ dst, int n) {
  int i = (blockIdx.x * blockDim.x + threadIdx.x) * 8;
  if (i < n) {
    float4 f0 = *(const float4*)(src + i);
    float4 f1 = *(const float4*)(src + i + 4);
    uint4 p;
    p.x = pack_bf16x2(f0.x, f0.y);
    p.y = pack_bf16x2(f0.z, f0.w);
    p.z = pack_bf16x2(f1.x, f1.y);
    p.w = pack_bf16x2(f1.z, f1.w);
    *(uint4*)(dst + i) = p;
  }
}

// ---------------------------------------------------------------------------
// Transpose f32 [K,N] -> bf16 [N,K] via padded LDS tile (coalesced both ways).
// block (32,8), grid (N/32, K/32).
// ---------------------------------------------------------------------------
__global__ void k_transpose_to_bf16(const float* __restrict__ src,
                                    unsigned short* __restrict__ dst,
                                    int K, int N) {
  __shared__ float tile[32][33];
  const int nb = blockIdx.x * 32;   // source column block
  const int kb = blockIdx.y * 32;   // source row block
  const int tx = threadIdx.x;
  const int ty = threadIdx.y;
#pragma unroll
  for (int r = 0; r < 32; r += 8)
    tile[ty + r][tx] = src[(size_t)(kb + ty + r) * N + (nb + tx)];
  __syncthreads();
#pragma unroll
  for (int r = 0; r < 32; r += 8)
    dst[(size_t)(nb + ty + r) * K + (kb + tx)] = f32_to_bf16_rne(tile[tx][ty + r]);
}

__global__ void k_zero_bf16(unsigned short* __restrict__ p, int n) {
  int i = blockIdx.x * blockDim.x + threadIdx.x;
  if (i < n) p[i] = 0;
}

// ---------------------------------------------------------------------------
// Fragment layouts (v_wmma_f32_16x16x32_bf16, wave32):
// A (16x32): lane<16 row=lane,  K in {k..k+15}
//            lane>=16 row=l-16, K in {k+16..k+31}
// B (32x16): lane<16 col=lane,  K in {k..k+15};  lane>=16 col=l-16, K in {k+16..k+31}
// C/D:       VGPR r -> row = r + (lane<16 ? 0 : 8), col = lane&15
// ---------------------------------------------------------------------------

// Phase 1: C[m,n] = sum_k A[m,k]*BT[n,k] + bias[n].
// One 16-row M-tile x four 16-col N-tiles per wave: A fragment loaded once,
// feeds 4 WMMAs per k-step (4x less A traffic, 4 wmma per 10 b128 loads).
__global__ void k_xw_gemm(const unsigned short* __restrict__ A,   // [M, IDIM] bf16
                          const unsigned short* __restrict__ BT,  // [HDIM, IDIM] bf16 (W_xh^T)
                          const float* __restrict__ bias,         // [HDIM]
                          float* __restrict__ C) {                // [M, HDIM]
  const int lane  = threadIdx.x & 31;
  const int wave  = blockIdx.x * (blockDim.x >> 5) + (threadIdx.x >> 5);
  const int g     = wave & 15;          // 16 groups of 4 N-tiles (HDIM/64)
  const int tileM = wave >> 4;
  const int row   = tileM * 16 + (lane & 15);
  const int col0  = g * 64 + (lane & 15);
  const int kA    = (lane < 16) ? 0 : 16;
  const int kB    = (lane < 16) ? 0 : 16;

  const unsigned short* aRow = A + (size_t)row * IDIM;
  const unsigned short* bRow0 = BT + (size_t)(col0 +  0) * IDIM;
  const unsigned short* bRow1 = BT + (size_t)(col0 + 16) * IDIM;
  const unsigned short* bRow2 = BT + (size_t)(col0 + 32) * IDIM;
  const unsigned short* bRow3 = BT + (size_t)(col0 + 48) * IDIM;

  v8f c0, c1, c2, c3;
  {
    float b0 = bias[col0], b1 = bias[col0 + 16];
    float b2 = bias[col0 + 32], b3 = bias[col0 + 48];
#pragma unroll
    for (int r = 0; r < 8; ++r) { c0[r] = b0; c1[r] = b1; c2[r] = b2; c3[r] = b3; }
  }

#pragma unroll 2
  for (int k = 0; k < IDIM; k += 32) {
    Frag16 a, b0, b1, b2, b3;
    a.q[0]  = *(const uint4*)(aRow  + k + kA);
    a.q[1]  = *(const uint4*)(aRow  + k + kA + 8);
    b0.q[0] = *(const uint4*)(bRow0 + k + kB);
    b0.q[1] = *(const uint4*)(bRow0 + k + kB + 8);
    b1.q[0] = *(const uint4*)(bRow1 + k + kB);
    b1.q[1] = *(const uint4*)(bRow1 + k + kB + 8);
    b2.q[0] = *(const uint4*)(bRow2 + k + kB);
    b2.q[1] = *(const uint4*)(bRow2 + k + kB + 8);
    b3.q[0] = *(const uint4*)(bRow3 + k + kB);
    b3.q[1] = *(const uint4*)(bRow3 + k + kB + 8);
    c0 = __builtin_amdgcn_wmma_f32_16x16x32_bf16(false, a.v, false, b0.v, (short)0, c0, false, false);
    c1 = __builtin_amdgcn_wmma_f32_16x16x32_bf16(false, a.v, false, b1.v, (short)0, c1, false, false);
    c2 = __builtin_amdgcn_wmma_f32_16x16x32_bf16(false, a.v, false, b2.v, (short)0, c2, false, false);
    c3 = __builtin_amdgcn_wmma_f32_16x16x32_bf16(false, a.v, false, b3.v, (short)0, c3, false, false);
  }

  const int rbase = (lane < 16) ? 0 : 8;
  float* cptr = C + (size_t)(tileM * 16 + rbase) * HDIM + col0;
#pragma unroll
  for (int r = 0; r < 8; ++r) {
    cptr[(size_t)r * HDIM +  0] = c0[r];
    cptr[(size_t)r * HDIM + 16] = c1[r];
    cptr[(size_t)r * HDIM + 32] = c2[r];
    cptr[(size_t)r * HDIM + 48] = c3[r];
  }
}

// ---------------------------------------------------------------------------
// Phase 2, one timestep: h = tanh(XW_t + Hprev @ W_hh)
// XW_t lives in the output buffer and is overwritten in place with h (f32);
// h also stored as bf16 for the next step's A fragments.
// 256 waves = 4 M-tiles x 64 N-tiles (max parallelism on the critical path).
// ---------------------------------------------------------------------------
__global__ void k_rnn_step(const unsigned short* __restrict__ Hprev, // [B, H] bf16
                           const unsigned short* __restrict__ WhhT,  // [H, H] bf16 (W_hh^T)
                           float* XWOut,                             // [B, H] f32: read xW, write h
                           unsigned short* __restrict__ Hnext,       // [B, H] bf16
                           float* __restrict__ LastState) {          // [B, H] or nullptr
  const int lane  = threadIdx.x & 31;
  const int wave  = blockIdx.x * (blockDim.x >> 5) + (threadIdx.x >> 5);
  const int tileN = wave & 63;
  const int tileM = wave >> 6;          // 0..3
  const int row   = tileM * 16 + (lane & 15);
  const int col   = tileN * 16 + (lane & 15);
  const int kA    = (lane < 16) ? 0 : 16;
  const int kB    = (lane < 16) ? 0 : 16;
  const int rbase = (lane < 16) ? 0 : 8;

  const unsigned short* aRow = Hprev + (size_t)row * HDIM;
  const unsigned short* bRow = WhhT  + (size_t)col * HDIM;

  // accumulator starts at xW_t (bias already folded in by phase 1)
  float* xptr = XWOut + (size_t)(tileM * 16 + rbase) * HDIM + col;
  v8f c;
#pragma unroll
  for (int r = 0; r < 8; ++r) c[r] = xptr[(size_t)r * HDIM];

#pragma unroll 4
  for (int k = 0; k < HDIM; k += 32) {
    __builtin_prefetch(bRow + k + 256, 0, 3);   // near-cache prefetch of W_hh slice
    Frag16 a, b;
    a.q[0] = *(const uint4*)(aRow + k + kA);
    a.q[1] = *(const uint4*)(aRow + k + kA + 8);
    b.q[0] = *(const uint4*)(bRow + k + kB);
    b.q[1] = *(const uint4*)(bRow + k + kB + 8);
    c = __builtin_amdgcn_wmma_f32_16x16x32_bf16(
        false, a.v, false, b.v, (short)0, c, false, false);
  }

  unsigned short* hptr = Hnext + (size_t)(tileM * 16 + rbase) * HDIM + col;
  float* lptr = LastState ? (LastState + (size_t)(tileM * 16 + rbase) * HDIM + col)
                          : nullptr;
#pragma unroll
  for (int r = 0; r < 8; ++r) {
    float h = tanhf(c[r]);
    xptr[(size_t)r * HDIM] = h;                  // outputs[t]
    hptr[(size_t)r * HDIM] = f32_to_bf16_rne(h); // state for t+1
    if (lptr) lptr[(size_t)r * HDIM] = h;        // final state
  }
}

// ---------------------------------------------------------------------------
extern "C" void kernel_launch(void* const* d_in, const int* in_sizes, int n_in,
                              void* d_out, int out_size, void* d_ws, size_t ws_size,
                              hipStream_t stream) {
  const float* x   = (const float*)d_in[0];  // [T, B, I]
  const float* Wxh = (const float*)d_in[1];  // [I, H]
  const float* Whh = (const float*)d_in[2];  // [H, H]
  const float* bh  = (const float*)d_in[3];  // [H]
  float* out = (float*)d_out;                // [T*B*H] outputs ++ [B*H] last state

  // workspace: x_bf16(64MB) | WxhT(2MB) | WhhT(2MB) | hbuf0(128KB) | hbuf1(128KB)
  char* ws = (char*)d_ws;
  unsigned short* x_bf = (unsigned short*)ws;
  unsigned short* WxhT = x_bf + (size_t)T_STEPS * BATCH * IDIM;
  unsigned short* WhhT = WxhT + (size_t)IDIM * HDIM;
  unsigned short* hb0  = WhhT + (size_t)HDIM * HDIM;
  unsigned short* hb1  = hb0 + (size_t)BATCH * HDIM;
  unsigned short* hb[2] = {hb0, hb1};

  // 0) convert inputs to bf16 (streaming), pre-transpose weights to bf16 [N,K]
  {
    int n = T_STEPS * BATCH * IDIM;          // 33,554,432
    k_f32_to_bf16<<<n / 8 / 256, 256, 0, stream>>>(x, x_bf, n);
  }
  dim3 tb(32, 8);
  k_transpose_to_bf16<<<dim3(HDIM / 32, IDIM / 32), tb, 0, stream>>>(Wxh, WxhT, IDIM, HDIM);
  k_transpose_to_bf16<<<dim3(HDIM / 32, HDIM / 32), tb, 0, stream>>>(Whh, WhhT, HDIM, HDIM);

  // 1) h_{-1} = 0
  k_zero_bf16<<<(BATCH * HDIM) / 256, 256, 0, stream>>>(hb0, BATCH * HDIM);

  // 2) xW = x @ W_xh + b_h, written into the outputs region
  //    waves = (T*B/16) M-tiles * (H/64) N-groups = 2048*16 = 32768; 8 waves/block
  k_xw_gemm<<<32768 / 8, 256, 0, stream>>>(x_bf, WxhT, bh, out);

  // 3) recurrence: 512 dependent step kernels (graph-captured)
  float* last = out + (size_t)T_STEPS * BATCH * HDIM;
  for (int t = 0; t < T_STEPS; ++t) {
    float* xw_t = out + (size_t)t * BATCH * HDIM;
    k_rnn_step<<<32, 256, 0, stream>>>(hb[t & 1], WhhT, xw_t, hb[(t + 1) & 1],
                                       (t == T_STEPS - 1) ? last : nullptr);
  }
}
